// SlidingTileAttention2D_46548855554077
// MI455X (gfx1250) — matrix operand order repaired and checked by reading, not compiled, not run on hardware
//
#include <hip/hip_runtime.h>
#include <hip/hip_bf16.h>

typedef __bf16 bf16;
typedef bf16  v16bf __attribute__((ext_vector_type(16)));
typedef bf16  v8bf  __attribute__((ext_vector_type(8)));
typedef float v8f   __attribute__((ext_vector_type(8)));
typedef unsigned int u32x4 __attribute__((ext_vector_type(4)));
typedef int i32x8 __attribute__((ext_vector_type(8)));
typedef int i32x4 __attribute__((ext_vector_type(4)));

// ---------------------------------------------------------------------------
// WMMA fragment loaders (CDNA5 ISA 7.12.2 layouts, wave32)
// ---------------------------------------------------------------------------
__device__ __forceinline__ v16bf load_frag_a(const bf16* base, int stride, int lane) {
  int row = lane & 15;
  int k0  = (lane & 16) ? 8 : 0;
  const bf16* p = base + row * stride + k0;
  v8bf lo = *(const v8bf*)(p);
  v8bf hi = *(const v8bf*)(p + 16);
  return __builtin_shufflevector(lo, hi, 0,1,2,3,4,5,6,7,8,9,10,11,12,13,14,15);
}

__device__ __forceinline__ v16bf load_frag_b(const bf16* base, int stride, int lane) {
  int n  = lane & 15;
  int k0 = (lane & 16) ? 8 : 0;
  const bf16* p = base + n * stride + k0;
  v8bf lo = *(const v8bf*)(p);
  v8bf hi = *(const v8bf*)(p + 16);
  return __builtin_shufflevector(lo, hi, 0,1,2,3,4,5,6,7,8,9,10,11,12,13,14,15);
}

__device__ __forceinline__ v8f wmma_bf16(v16bf a, v16bf b, v8f c) {
  return __builtin_amdgcn_wmma_f32_16x16x32_bf16(false, a, false, b, (short)0, c,
                                                 false, false);
}

// ---------------------------------------------------------------------------
// Tensor Data Mover: 2-D bf16 tile, global -> LDS, with LDS row padding.
// D# packing per CDNA5 ISA 8.3/8.4 (count=1, type=2, data_size=2B, pad enable).
// pad_int_code: DWORDs-per-row code (3 -> 16 DW = 32 bf16, 4 -> 32 DW = 64 bf16)
// pad_amt_dw:   pad DWORDs appended per row (field stores value-1)
// Uses the 6-arg clang-23 builtin form (extra zero int32x8 group + cpol).
// ---------------------------------------------------------------------------
__device__ __forceinline__ void tdm_load_2d(const void* lds_dst, const void* gsrc,
                                            int tile_w, int tile_h, int g_stride,
                                            int pad_int_code, int pad_amt_dw) {
  unsigned long long ga = (unsigned long long)(size_t)gsrc;
  unsigned la = (unsigned)(size_t)lds_dst;  // LDS aperture keeps offset in [31:0]
  u32x4 g0;
  g0[0] = 1u;                                    // count=1 (valid user D#)
  g0[1] = la;                                    // lds_addr
  g0[2] = (unsigned)ga;                          // global_addr[31:0]
  g0[3] = (unsigned)(ga >> 32) | (2u << 30);     // global_addr[56:32] | type=2
  i32x8 g1;
  g1[0] = (1 << 16) | (1 << 20) | (pad_int_code << 22) | ((pad_amt_dw - 1) << 25);
  g1[1] = (g_stride & 0xffff) << 16;             // tensor_dim0[15:0]
  g1[2] = (g_stride >> 16) | ((tile_h & 0xffff) << 16);   // dim0 hi | dim1 lo
  g1[3] = (tile_h >> 16) | (tile_w << 16);       // dim1 hi | tile_dim0
  g1[4] = tile_h;                                 // tile_dim1 (tile_dim2 = 0)
  g1[5] = g_stride;                               // tensor_dim0_stride[31:0]
  g1[6] = 0;
  g1[7] = 0;
  i32x4 gz;
  gz[0] = 0; gz[1] = 0; gz[2] = 0; gz[3] = 0;
  i32x8 gz8;
  gz8[0] = 0; gz8[1] = 0; gz8[2] = 0; gz8[3] = 0;
  gz8[4] = 0; gz8[5] = 0; gz8[6] = 0; gz8[7] = 0;
  __builtin_amdgcn_tensor_load_to_lds(g0, g1, gz, gz, gz8, 0);
}

// ---------------------------------------------------------------------------
// Kernel 0a: f32 -> bf16 (hidden states, row-major kept)
// ---------------------------------------------------------------------------
__global__ __launch_bounds__(256) void k_cvt(bf16* __restrict__ dst,
                                             const float* __restrict__ src) {
  size_t i = ((size_t)blockIdx.x * 256 + threadIdx.x) * 4;
  float4 f = *(const float4*)(src + i);
  dst[i + 0] = (bf16)f.x;
  dst[i + 1] = (bf16)f.y;
  dst[i + 2] = (bf16)f.z;
  dst[i + 3] = (bf16)f.w;
}

// ---------------------------------------------------------------------------
// Kernel 0b: f32 [R,C] -> bf16 transposed [C,R] (weights, N-major for B-frags)
// ---------------------------------------------------------------------------
__global__ __launch_bounds__(256) void k_tcvt(bf16* __restrict__ dst,
                                              const float* __restrict__ src,
                                              int R, int C) {
  int idx = blockIdx.x * 256 + threadIdx.x;
  int r = idx / C, c = idx % C;
  dst[(size_t)c * R + r] = (bf16)src[(size_t)r * C + c];
}

// ---------------------------------------------------------------------------
// Kernel 1: fused QKV projection. M=8192, N=2048, K=2048, blockIdx.z = weight.
// TDM double-buffered staging; epilogue scatters into tile order.
// ---------------------------------------------------------------------------
__global__ __launch_bounds__(256) void k_gemm_qkv(
    const bf16* __restrict__ hsb,
    const bf16* __restrict__ WqT, const bf16* __restrict__ WkT,
    const bf16* __restrict__ WvT,
    bf16* __restrict__ qb, bf16* __restrict__ kb, bf16* __restrict__ vtb) {
  __shared__ bf16 As[2][128 * 40];
  __shared__ bf16 Bs[2][128 * 40];
  const int tid = threadIdx.x;
  const int lane = tid & 31, w = tid >> 5;
  const int wm = w >> 1, wn = w & 1;
  const int half = lane >> 4, ln = lane & 15;
  const int gm0 = blockIdx.y * 128, gn0 = blockIdx.x * 128;
  const int z = blockIdx.z;
  const bf16* WT = (z == 0) ? WqT : (z == 1) ? WkT : WvT;
  const bf16* Abase = hsb + (size_t)gm0 * 2048;
  const bf16* Bbase = WT + (size_t)gn0 * 2048;

  v8f acc[2][4];
#pragma unroll
  for (int mt = 0; mt < 2; mt++)
#pragma unroll
    for (int nt = 0; nt < 4; nt++) acc[mt][nt] = (v8f){0,0,0,0,0,0,0,0};

  if (w == 0) {  // prologue: DMA tile 0
    tdm_load_2d(&As[0][0], Abase, 32, 128, 2048, 3, 4);
    tdm_load_2d(&Bs[0][0], Bbase, 32, 128, 2048, 3, 4);
  }

  for (int i = 0; i < 64; i++) {
    int cur = i & 1;
    if (w == 0) __builtin_amdgcn_s_wait_tensorcnt(0);
    __syncthreads();
    if (w == 0 && i < 63) {  // DMA next tile into other buffer
      tdm_load_2d(&As[cur ^ 1][0], Abase + (i + 1) * 32, 32, 128, 2048, 3, 4);
      tdm_load_2d(&Bs[cur ^ 1][0], Bbase + (i + 1) * 32, 32, 128, 2048, 3, 4);
    }
    v16bf af[2], bfr[4];
#pragma unroll
    for (int mt = 0; mt < 2; mt++)
      af[mt] = load_frag_a(&As[cur][(wm * 32 + mt * 16) * 40], 40, lane);
#pragma unroll
    for (int nt = 0; nt < 4; nt++)
      bfr[nt] = load_frag_b(&Bs[cur][(wn * 64 + nt * 16) * 40], 40, lane);
#pragma unroll
    for (int mt = 0; mt < 2; mt++)
#pragma unroll
      for (int nt = 0; nt < 4; nt++)
        acc[mt][nt] = wmma_bf16(af[mt], bfr[nt], acc[mt][nt]);
  }

#pragma unroll
  for (int mt = 0; mt < 2; mt++)
#pragma unroll
    for (int nt = 0; nt < 4; nt++)
#pragma unroll
      for (int r = 0; r < 8; r++) {
        int m = gm0 + wm * 32 + mt * 16 + r + half * 8;
        int n = gn0 + wn * 64 + nt * 16 + ln;
        int b = m >> 12, s = m & 4095;
        int h = n >> 6, d = n & 63;
        int nth = s >> 9, th = (s >> 6) & 7, ntw = (s >> 4) & 3, tw = s & 15;
        int tt = nth * 4 + ntw, j = th * 16 + tw;
        bf16 v = (bf16)acc[mt][nt][r];
        size_t hb = ((size_t)b * 32 + h) * 32 + tt;
        if (z == 0)      qb[hb * 8192 + j * 64 + d] = v;
        else if (z == 1) kb[hb * 8192 + j * 64 + d] = v;
        else             vtb[hb * 8192 + d * 128 + j] = v;
      }
}

// ---------------------------------------------------------------------------
// Kernel 2: sliding-tile flash attention. One block per (b,h,tile). 8 waves x
// 16 query rows. 8 kv half-chunks of 64 tokens staged by TDM.
// ---------------------------------------------------------------------------
__global__ __launch_bounds__(256) void k_attn(const bf16* __restrict__ qb,
                                              const bf16* __restrict__ kb,
                                              const bf16* __restrict__ vtb,
                                              bf16* __restrict__ ob) {
  __shared__ bf16 Qs[128 * 72];
  __shared__ bf16 Ks[64 * 72];
  __shared__ bf16 Vts[64 * 72];
  __shared__ bf16 Ps[128 * 72];
  __shared__ float m_s[128];
  __shared__ float l_s[128];

  const int tid = threadIdx.x;
  const int lane = tid & 31, w = tid >> 5;
  const int half = lane >> 4, ln = lane & 15;
  int idx = blockIdx.x;
  int b = idx >> 10, rem = idx & 1023;
  int h = rem >> 5, t = rem & 31;

  // static 2x2 sliding tile window (wh=ww=2, Ht=8, Wt=4)
  int tr = t >> 2, tc = t & 3;
  int cr = min(max(tr, 1), 7);
  int cc = min(max(tc, 1), 3);
  int kts[4] = {(cr - 1) * 4 + (cc - 1), (cr - 1) * 4 + cc,
                cr * 4 + (cc - 1),       cr * 4 + cc};

  size_t base = ((size_t)b * 32 + h) * 32;

  if (w == 0)  // Q tile 128x64 via TDM (rows 64 bf16 -> pad to 72)
    tdm_load_2d(Qs, qb + (base + t) * 8192, 64, 128, 64, 4, 4);
  if (tid < 128) { m_s[tid] = -1e30f; l_s[tid] = 0.f; }

  v8f o[4];
#pragma unroll
  for (int nt = 0; nt < 4; nt++) o[nt] = (v8f){0,0,0,0,0,0,0,0};

  for (int c = 0; c < 8; c++) {
    __syncthreads();  // all waves done reading previous Ks/Vts
    if (w == 0) {
      int kt = kts[c >> 1], coff = (c & 1) * 64;
      tdm_load_2d(Ks, kb + (base + kt) * 8192 + coff * 64, 64, 64, 64, 4, 4);
      tdm_load_2d(Vts, vtb + (base + kt) * 8192 + coff, 64, 64, 128, 4, 4);
      __builtin_amdgcn_s_wait_tensorcnt(0);
    }
    __syncthreads();

    // S = Q * K^T  (wave's 16 rows x 64 kv cols)
    v8f sc[4];
#pragma unroll
    for (int nt = 0; nt < 4; nt++) sc[nt] = (v8f){0,0,0,0,0,0,0,0};
#pragma unroll
    for (int kk = 0; kk < 64; kk += 32) {
      v16bf aq = load_frag_a(&Qs[(w * 16) * 72 + kk], 72, lane);
#pragma unroll
      for (int nt = 0; nt < 4; nt++) {
        v16bf bk = load_frag_b(&Ks[(nt * 16) * 72 + kk], 72, lane);
        sc[nt] = wmma_bf16(aq, bk, sc[nt]);
      }
    }

    // online softmax (C layout: VGPR r -> row r/r+8; 16-lane shfl reductions)
    float alpha[8];
#pragma unroll
    for (int r = 0; r < 8; r++) {
      int R = w * 16 + r + half * 8;
      float mx = -1e30f;
#pragma unroll
      for (int nt = 0; nt < 4; nt++) {
        float sv = sc[nt][r] * 0.125f;  // 1/sqrt(64)
        sc[nt][r] = sv;
        mx = fmaxf(mx, sv);
      }
#pragma unroll
      for (int mm = 8; mm >= 1; mm >>= 1) mx = fmaxf(mx, __shfl_xor(mx, mm, 32));
      float mo = m_s[R];
      float mn = fmaxf(mo, mx);
      float a = __expf(mo - mn);
      float sum = 0.f;
#pragma unroll
      for (int nt = 0; nt < 4; nt++) {
        float p = __expf(sc[nt][r] - mn);
        sc[nt][r] = p;
        sum += p;
      }
#pragma unroll
      for (int mm = 8; mm >= 1; mm >>= 1) sum += __shfl_xor(sum, mm, 32);
      if (ln == 0) { l_s[R] = a * l_s[R] + sum; m_s[R] = mn; }
      alpha[r] = a;
    }
#pragma unroll
    for (int nt = 0; nt < 4; nt++)
#pragma unroll
      for (int r = 0; r < 8; r++) o[nt][r] *= alpha[r];

    // spill P (bf16) to LDS to re-layout C->A fragments (per-wave region)
#pragma unroll
    for (int nt = 0; nt < 4; nt++)
#pragma unroll
      for (int r = 0; r < 8; r++)
        Ps[(w * 16 + r + half * 8) * 72 + nt * 16 + ln] = (bf16)sc[nt][r];

    // O += P * V
#pragma unroll
    for (int kk = 0; kk < 64; kk += 32) {
      v16bf ap = load_frag_a(&Ps[(w * 16) * 72 + kk], 72, lane);
#pragma unroll
      for (int nt = 0; nt < 4; nt++) {
        v16bf bv = load_frag_b(&Vts[(nt * 16) * 72 + kk], 72, lane);
        o[nt] = wmma_bf16(ap, bv, o[nt]);
      }
    }
  }

  // normalize and scatter back to token order, bf16 for the output GEMM
  int nthi = t >> 2, ntwi = t & 3;
#pragma unroll
  for (int r = 0; r < 8; r++) {
    int R = w * 16 + r + half * 8;
    float inv = 1.0f / l_s[R];
    int th = R >> 4, tw = R & 15;
    int s = nthi * 512 + th * 64 + ntwi * 16 + tw;
    size_t ro = ((size_t)b * 4096 + s) * 2048 + h * 64;
#pragma unroll
    for (int nt = 0; nt < 4; nt++)
      ob[ro + nt * 16 + ln] = (bf16)(o[nt][r] * inv);
  }
}

// ---------------------------------------------------------------------------
// Kernel 3: output projection. A = attn out (bf16), B = WoT, C -> f32 d_out.
// ---------------------------------------------------------------------------
__global__ __launch_bounds__(256) void k_gemm_out(const bf16* __restrict__ Abf,
                                                  const bf16* __restrict__ WoT,
                                                  float* __restrict__ out) {
  __shared__ bf16 As[2][128 * 40];
  __shared__ bf16 Bs[2][128 * 40];
  const int tid = threadIdx.x;
  const int lane = tid & 31, w = tid >> 5;
  const int wm = w >> 1, wn = w & 1;
  const int half = lane >> 4, ln = lane & 15;
  const int gm0 = blockIdx.y * 128, gn0 = blockIdx.x * 128;
  const bf16* Abase = Abf + (size_t)gm0 * 2048;
  const bf16* Bbase = WoT + (size_t)gn0 * 2048;

  v8f acc[2][4];
#pragma unroll
  for (int mt = 0; mt < 2; mt++)
#pragma unroll
    for (int nt = 0; nt < 4; nt++) acc[mt][nt] = (v8f){0,0,0,0,0,0,0,0};

  if (w == 0) {
    tdm_load_2d(&As[0][0], Abase, 32, 128, 2048, 3, 4);
    tdm_load_2d(&Bs[0][0], Bbase, 32, 128, 2048, 3, 4);
  }

  for (int i = 0; i < 64; i++) {
    int cur = i & 1;
    if (w == 0) __builtin_amdgcn_s_wait_tensorcnt(0);
    __syncthreads();
    if (w == 0 && i < 63) {
      tdm_load_2d(&As[cur ^ 1][0], Abase + (i + 1) * 32, 32, 128, 2048, 3, 4);
      tdm_load_2d(&Bs[cur ^ 1][0], Bbase + (i + 1) * 32, 32, 128, 2048, 3, 4);
    }
    v16bf af[2], bfr[4];
#pragma unroll
    for (int mt = 0; mt < 2; mt++)
      af[mt] = load_frag_a(&As[cur][(wm * 32 + mt * 16) * 40], 40, lane);
#pragma unroll
    for (int nt = 0; nt < 4; nt++)
      bfr[nt] = load_frag_b(&Bs[cur][(wn * 64 + nt * 16) * 40], 40, lane);
#pragma unroll
    for (int mt = 0; mt < 2; mt++)
#pragma unroll
      for (int nt = 0; nt < 4; nt++)
        acc[mt][nt] = wmma_bf16(af[mt], bfr[nt], acc[mt][nt]);
  }

#pragma unroll
  for (int mt = 0; mt < 2; mt++)
#pragma unroll
    for (int nt = 0; nt < 4; nt++)
#pragma unroll
      for (int r = 0; r < 8; r++) {
        int m = gm0 + wm * 32 + mt * 16 + r + half * 8;
        int n = gn0 + wn * 64 + nt * 16 + ln;
        out[(size_t)m * 2048 + n] = acc[mt][nt][r];
      }
}

// ---------------------------------------------------------------------------
extern "C" void kernel_launch(void* const* d_in, const int* in_sizes, int n_in,
                              void* d_out, int out_size, void* d_ws,
                              size_t ws_size, hipStream_t stream) {
  (void)in_sizes; (void)n_in; (void)out_size; (void)ws_size;
  const float* hidden = (const float*)d_in[0];
  const float* Wq = (const float*)d_in[1];
  const float* Wk = (const float*)d_in[2];
  const float* Wv = (const float*)d_in[3];
  const float* Wo = (const float*)d_in[4];

  const size_t WT_SZ = 2048ull * 2048ull;          // elems per weight
  const size_t TBUF  = 2ull * 32 * 32 * 128 * 64;  // elems per big buffer
  bf16* WqT = (bf16*)d_ws;
  bf16* WkT = WqT + WT_SZ;
  bf16* WvT = WkT + WT_SZ;
  bf16* WoT = WvT + WT_SZ;
  bf16* hsb = WoT + WT_SZ;   // bf16 hidden; dead after QKV GEMM
  bf16* obf = hsb;           // attention output aliases it (launch-ordered)
  bf16* qb  = hsb + TBUF;
  bf16* kb  = qb + TBUF;
  bf16* vtb = kb + TBUF;

  // conversions: hidden -> bf16; weights -> bf16 transposed (N-major)
  k_cvt<<<16384, 256, 0, stream>>>(hsb, hidden);
  k_tcvt<<<16384, 256, 0, stream>>>(WqT, Wq, 2048, 2048);
  k_tcvt<<<16384, 256, 0, stream>>>(WkT, Wk, 2048, 2048);
  k_tcvt<<<16384, 256, 0, stream>>>(WvT, Wv, 2048, 2048);
  k_tcvt<<<16384, 256, 0, stream>>>(WoT, Wo, 2048, 2048);

  // fused QKV projection into tile-ordered buffers (V transposed)
  k_gemm_qkv<<<dim3(16, 64, 3), 256, 0, stream>>>(hsb, WqT, WkT, WvT, qb, kb,
                                                  vtb);

  // sliding-tile flash attention: one block per (b, head, query tile)
  k_attn<<<2048, 256, 0, stream>>>(qb, kb, vtb, obf);

  // output projection -> f32
  k_gemm_out<<<dim3(16, 64), 256, 0, stream>>>(obf, WoT, (float*)d_out);
}
